// AutoregLstm_22548578304213
// MI455X (gfx1250) — matrix operand order repaired and checked by reading, not compiled, not run on hardware
//
#include <hip/hip_runtime.h>
#include <math.h>

// Problem dims (fixed by the reference)
#define B_      512
#define T_      512
#define H_      256
#define OUTL_   8
#define DIN_    64
#define DOUT_   64

typedef __attribute__((ext_vector_type(16))) _Float16 v16h;
typedef __attribute__((ext_vector_type(8)))  _Float16 v8h;
typedef __attribute__((ext_vector_type(4)))  _Float16 v4h;
typedef __attribute__((ext_vector_type(8)))  float    v8f;

// ---------------------------------------------------------------------------
// Generic WMMA GEMM:  C[M,N] = act( A*B (+ C if accum) + bias )
//   A: [M,K] row-major, row stride lda (lets us read [B,T,H] time-slices)
//   B: transB ? stored [N,K] (ldb = K-stride)  : stored [K,N] (ldb = N-stride)
//   Requires M%128==0, N%64==0, K%32==0, all row strides %4 (true here).
// Workgroup = 128 threads = 4 waves; block tile 128x64; each wave computes a
// 32x64 strip: 2 A-frags x 4 B-frags -> 8 v_wmma per 32-deep K step, i.e.
// 12 ds_load_b128 feed 8 WMMAs. Staging is split into an issue phase (all
// global_load_b128 in flight) and a drain phase (cvt + ds_store) so the
// compiler emits batched loads with descending partial loadcnt waits.
// f32 -> f16 conversion during LDS staging; f32 accumulation.
// B is staged directly in WMMA B-fragment order so fragment loads are
// contiguous ds_load_b128 per lane (no per-element column gathers).
// ---------------------------------------------------------------------------
__global__ __launch_bounds__(128)
void gemm_f16_wmma(const float* __restrict__ A, int lda,
                   const float* __restrict__ Bm, int ldb, int transB,
                   const float* __restrict__ bias,
                   float* __restrict__ C, int ldc,
                   int M, int N, int K, int accum, int relu)
{
    // A tile 128x32, pitch 40 halves (80 B) keeps 16B alignment of frag slices
    __shared__ _Float16 As[128][40];
    // B fragments: [nGroup 0..3][wmma lane][k 0..15], padded to 24 halves
    // (48 B lane stride: 16B-aligned b128 loads, benign bank pattern)
    __shared__ _Float16 Bf[4][32][24];

    const int tid  = threadIdx.x;
    const int wave = tid >> 5;
    const int lane = tid & 31;
    const int wr   = wave << 5;             // wave row offset in 128-row tile
    const int lrow = lane & 15;
    const int lhi  = lane >> 4;             // half-wave select
    const int brow = blockIdx.y * 128;
    const int bcol = blockIdx.x * 64;

    v8f acc[2][4];
    #pragma unroll
    for (int i = 0; i < 2; ++i)
        #pragma unroll
        for (int j = 0; j < 4; ++j)
            acc[i][j] = (v8f){};

    for (int k0 = 0; k0 < K; k0 += 32) {
        float4 a4[8], b4[4];
        // ---- issue all global loads (A 128x32, B 32x64) ----
        #pragma unroll
        for (int it = 0; it < 8; ++it) {
            int idx = it * 512 + tid * 4;
            int r = idx >> 5, c = idx & 31;
            a4[it] = *(const float4*)(A + (size_t)(brow + r) * lda + (k0 + c));
        }
        if (transB) {
            #pragma unroll
            for (int it = 0; it < 4; ++it) {
                int ch = it * 128 + tid;
                int n = ch >> 3, kc = (ch & 7) * 4;
                b4[it] = *(const float4*)(Bm + (size_t)(bcol + n) * ldb + (k0 + kc));
            }
        } else {
            #pragma unroll
            for (int it = 0; it < 4; ++it) {
                int ch = it * 128 + tid;
                int k = ch >> 4, n4 = (ch & 15) * 4;
                b4[it] = *(const float4*)(Bm + (size_t)(k0 + k) * ldb + (bcol + n4));
            }
        }
        // ---- drain: convert f32->f16 and store to LDS ----
        #pragma unroll
        for (int it = 0; it < 8; ++it) {
            int idx = it * 512 + tid * 4;
            int r = idx >> 5, c = idx & 31;
            v4h hv = { (_Float16)a4[it].x, (_Float16)a4[it].y,
                       (_Float16)a4[it].z, (_Float16)a4[it].w };
            *(v4h*)(&As[r][c]) = hv;
        }
        if (transB) {
            // fragment layout store: one b64 along K
            #pragma unroll
            for (int it = 0; it < 4; ++it) {
                int ch = it * 128 + tid;
                int n = ch >> 3, kc = (ch & 7) * 4;
                v4h hv = { (_Float16)b4[it].x, (_Float16)b4[it].y,
                           (_Float16)b4[it].z, (_Float16)b4[it].w };
                int ln = ((kc >> 4) << 4) | (n & 15);
                *(v4h*)(&Bf[n >> 4][ln][kc & 15]) = hv;
            }
        } else {
            // fragment layout store: scatter 4 b16 across lanes
            #pragma unroll
            for (int it = 0; it < 4; ++it) {
                int ch = it * 128 + tid;
                int k = ch >> 4, n4 = (ch & 15) * 4;
                int g   = n4 >> 4;
                int ln0 = (k >> 4) << 4;
                int nb  = n4 & 15;
                int i   = k & 15;
                Bf[g][ln0 | (nb + 0)][i] = (_Float16)b4[it].x;
                Bf[g][ln0 | (nb + 1)][i] = (_Float16)b4[it].y;
                Bf[g][ln0 | (nb + 2)][i] = (_Float16)b4[it].z;
                Bf[g][ln0 | (nb + 3)][i] = (_Float16)b4[it].w;
            }
        }
        __syncthreads();

        // ---- fragments: 2x b128 per fragment, concat via shufflevector ----
        v16h a[2], b[4];
        #pragma unroll
        for (int i = 0; i < 2; ++i) {
            const v8h lo = *(const v8h*)(&As[wr + i * 16 + lrow][lhi * 16]);
            const v8h hi = *(const v8h*)(&As[wr + i * 16 + lrow][lhi * 16 + 8]);
            a[i] = __builtin_shufflevector(lo, hi, 0,1,2,3,4,5,6,7,8,9,10,11,12,13,14,15);
        }
        #pragma unroll
        for (int j = 0; j < 4; ++j) {
            const v8h lo = *(const v8h*)(&Bf[j][lane][0]);
            const v8h hi = *(const v8h*)(&Bf[j][lane][8]);
            b[j] = __builtin_shufflevector(lo, hi, 0,1,2,3,4,5,6,7,8,9,10,11,12,13,14,15);
        }
        #pragma unroll
        for (int i = 0; i < 2; ++i)
            #pragma unroll
            for (int j = 0; j < 4; ++j)
                acc[i][j] = __builtin_amdgcn_wmma_f32_16x16x32_f16(
                    false, a[i], false, b[j], (short)0, acc[i][j], false, false);
        __syncthreads();
    }

    // ---- epilogue: bias / accumulate / ReLU (coalesced b32 across lanes) ----
    #pragma unroll
    for (int j = 0; j < 4; ++j) {
        int n = bcol + j * 16 + lrow;
        float bv = bias ? bias[n] : 0.0f;
        #pragma unroll
        for (int i = 0; i < 2; ++i) {
            #pragma unroll
            for (int e = 0; e < 8; ++e) {
                int m = brow + wr + i * 16 + lhi * 8 + e;
                size_t gi = (size_t)m * ldc + n;
                float v = acc[i][j][e] + bv;
                if (accum) v += C[gi];
                if (relu)  v = fmaxf(v, 0.0f);
                C[gi] = v;
            }
        }
    }
}

// ---------------------------------------------------------------------------
// Elementwise / reduction helpers
// ---------------------------------------------------------------------------
__device__ __forceinline__ float sigf(float x) { return 1.0f / (1.0f + __expf(-x)); }

__global__ void fill_f32(float* p, float v, int n) {
    int i = blockIdx.x * blockDim.x + threadIdx.x;
    if (i < n) p[i] = v;
}

__global__ void vec_add(const float* a, const float* b, float* o, int n) {
    int i = blockIdx.x * blockDim.x + threadIdx.x;
    if (i < n) o[i] = a[i] + b[i];
}

// Encoder LSTM cell: gates[B,4H] (i,f,g,o), c in/out, h -> enc[:,t,:] (row stride hstride)
__global__ void lstm_enc_cell(const float* __restrict__ g, float* __restrict__ c,
                              float* __restrict__ hdst, int hstride, int n)
{
    int idx = blockIdx.x * blockDim.x + threadIdx.x;
    if (idx >= n) return;
    int b = idx >> 8;           // H_ = 256
    int h = idx & 255;
    const float* gb = g + (size_t)b * 4 * H_;
    float gi = gb[h], gf = gb[H_ + h], gg = gb[2 * H_ + h], go = gb[3 * H_ + h];
    float cn = sigf(gf) * c[idx] + sigf(gi) * tanhf(gg);
    c[idx] = cn;
    hdst[(size_t)b * hstride + h] = sigf(go) * tanhf(cn);
}

// Decoder cell with fresh zero state: c = sig(i)*tanh(g); s = sig(o)*tanh(c)
__global__ void lstm_dec_cell(const float* __restrict__ g, float* __restrict__ s,
                              float* __restrict__ decseq, int step, int n)
{
    int idx = blockIdx.x * blockDim.x + threadIdx.x;
    if (idx >= n) return;
    int b = idx >> 8;
    int h = idx & 255;
    const float* gb = g + (size_t)b * 4 * H_;
    float gi = gb[h], gg = gb[2 * H_ + h], go = gb[3 * H_ + h];
    float cn = sigf(gi) * tanhf(gg);
    float hv = sigf(go) * tanhf(cn);
    s[idx] = hv;
    decseq[((size_t)b * OUTL_ + step) * H_ + h] = hv;
}

// z[b,t,h] = tanh(us[b,h] + whs[b,t,h])
__global__ void add_tanh(const float* __restrict__ us, const float* __restrict__ whs,
                         float* __restrict__ z)
{
    size_t i = (size_t)blockIdx.x * blockDim.x + threadIdx.x;
    int h = (int)(i & 255);
    size_t b = i >> 17;         // T_*H_ = 131072 = 2^17
    z[i] = tanhf(us[b * H_ + h] + whs[i]);
}

// e[row] = dot(z[row,:], w) + bl[0]   (the N=1 final v-layer)
__global__ __launch_bounds__(256)
void rowdot(const float* __restrict__ z, const float* __restrict__ w,
            const float* __restrict__ bl, float* __restrict__ e)
{
    __shared__ float red[256];
    int row = blockIdx.x, t = threadIdx.x;
    red[t] = z[(size_t)row * H_ + t] * w[t];
    __syncthreads();
    for (int s = 128; s > 0; s >>= 1) {
        if (t < s) red[t] += red[t + s];
        __syncthreads();
    }
    if (t == 0) e[row] = red[0] + bl[0];
}

// softmax over T per batch row (T=512, 256 threads, 2 elems/thread)
__global__ __launch_bounds__(256)
void softmax_T(const float* __restrict__ e, float* __restrict__ a)
{
    __shared__ float red[256];
    int b = blockIdx.x, t = threadIdx.x;
    float e0 = e[(size_t)b * T_ + t], e1 = e[(size_t)b * T_ + 256 + t];
    red[t] = fmaxf(e0, e1);
    __syncthreads();
    for (int s = 128; s > 0; s >>= 1) { if (t < s) red[t] = fmaxf(red[t], red[t + s]); __syncthreads(); }
    float mx = red[0];
    __syncthreads();
    float v0 = __expf(e0 - mx), v1 = __expf(e1 - mx);
    red[t] = v0 + v1;
    __syncthreads();
    for (int s = 128; s > 0; s >>= 1) { if (t < s) red[t] += red[t + s]; __syncthreads(); }
    float inv = 1.0f / red[0];
    a[(size_t)b * T_ + t] = v0 * inv;
    a[(size_t)b * T_ + 256 + t] = v1 * inv;
}

// c_t[b,h] = sum_t a[b,t] * enc[b,t,h]   (256 threads = h, coalesced over h)
__global__ __launch_bounds__(256)
void ctx_sum(const float* __restrict__ a, const float* __restrict__ enc,
             float* __restrict__ ct)
{
    int b = blockIdx.x, h = threadIdx.x;
    const float* eb = enc + (size_t)b * T_ * H_;
    const float* ab = a + (size_t)b * T_;
    float acc = 0.0f;
    for (int t = 0; t < T_; ++t) acc += ab[t] * eb[(size_t)t * H_ + h];
    ct[(size_t)b * H_ + h] = acc;
}

// ---------------------------------------------------------------------------
// Host orchestration
// ---------------------------------------------------------------------------
static inline void gemm(hipStream_t s, const float* A, int lda,
                        const float* Bm, int ldb, int tB, const float* bias,
                        float* C, int ldc, int M, int N, int K, int accum, int relu)
{
    dim3 g(N / 64, M / 128), b(128);
    gemm_f16_wmma<<<g, b, 0, s>>>(A, lda, Bm, ldb, tB, bias, C, ldc, M, N, K, accum, relu);
}

extern "C" void kernel_launch(void* const* d_in, const int* in_sizes, int n_in,
                              void* d_out, int out_size, void* d_ws, size_t ws_size,
                              hipStream_t stream)
{
    (void)in_sizes; (void)n_in; (void)out_size; (void)ws_size;
    const float* x        = (const float*)d_in[0];
    const float* in_W     = (const float*)d_in[1];
    const float* in_b     = (const float*)d_in[2];
    const float* enc_Wih  = (const float*)d_in[3];
    const float* enc_Whh  = (const float*)d_in[4];
    const float* enc_bih  = (const float*)d_in[5];
    const float* enc_bhh  = (const float*)d_in[6];
    const float* dec_Wih  = (const float*)d_in[7];
    /* dec_Whh (d_in[8]) never contributes: fresh zero state each step */
    const float* dec_bih  = (const float*)d_in[9];
    const float* dec_bhh  = (const float*)d_in[10];
    const float* w_Ws     = (const float*)d_in[11];
    const float* w_bs     = (const float*)d_in[12];
    const float* w_Wl     = (const float*)d_in[13];
    const float* w_bl     = (const float*)d_in[14];
    const float* u_Ws     = (const float*)d_in[15];
    const float* u_bs     = (const float*)d_in[16];
    const float* u_Wl     = (const float*)d_in[17];
    const float* u_bl     = (const float*)d_in[18];
    const float* v_Ws     = (const float*)d_in[19];
    const float* v_bs     = (const float*)d_in[20];
    const float* v_Wl     = (const float*)d_in[21];
    const float* v_bl     = (const float*)d_in[22];
    const float* out_W    = (const float*)d_in[23];
    const float* out_b    = (const float*)d_in[24];
    float* out = (float*)d_out;

    // workspace carve-up (floats)
    float* ws = (float*)d_ws;
    size_t off = 0;
    const size_t BT = (size_t)B_ * T_;
    float* bufh   = ws + off; off += BT * H_;              // input-proj h, then MLP pong
    float* enc    = ws + off; off += BT * H_;              // encoder outputs [B,T,H]
    float* whs    = ws + off; off += BT * H_;              // w(hs), loop-invariant
    float* buft   = ws + off; off += BT * H_;              // MLP ping
    float* gates  = ws + off; off += (size_t)B_ * 4 * H_;
    float* cstate = ws + off; off += (size_t)B_ * H_;
    float* sstate = ws + off; off += (size_t)B_ * H_;
    float* ct     = ws + off; off += (size_t)B_ * H_;
    float* usa    = ws + off; off += (size_t)B_ * H_;
    float* usb    = ws + off; off += (size_t)B_ * H_;
    float* evec   = ws + off; off += BT;
    float* avec   = ws + off; off += BT;
    float* decseq = ws + off; off += (size_t)B_ * OUTL_ * H_;
    float* bsum_e = ws + off; off += 4 * H_;
    float* bsum_d = ws + off; off += 4 * H_;

    const int nBH = B_ * H_;

    // combined biases, zero states (must re-init every call: ws is poisoned)
    vec_add<<<(4 * H_ + 255) / 256, 256, 0, stream>>>(enc_bih, enc_bhh, bsum_e, 4 * H_);
    vec_add<<<(4 * H_ + 255) / 256, 256, 0, stream>>>(dec_bih, dec_bhh, bsum_d, 4 * H_);
    fill_f32<<<(nBH + 255) / 256, 256, 0, stream>>>(cstate, 0.0f, nBH);
    fill_f32<<<(nBH + 255) / 256, 256, 0, stream>>>(sstate, 0.0f, nBH);

    // ---- input projection: h = x @ in_W + in_b  [BT,64]x[64,256] ----
    gemm(stream, x, DIN_, in_W, H_, 0, in_b, bufh, H_, (int)BT, H_, DIN_, 0, 0);

    // ---- encoder LSTM over T (sequential recurrence) ----
    for (int t = 0; t < T_; ++t) {
        // gates = x_t @ Wih^T + (bih+bhh);  x_t read in place from [B,T,H]
        gemm(stream, bufh + (size_t)t * H_, T_ * H_, enc_Wih, H_, 1, bsum_e,
             gates, 4 * H_, B_, 4 * H_, H_, 0, 0);
        if (t > 0)  // gates += h_{t-1} @ Whh^T; h_{t-1} read in place from enc
            gemm(stream, enc + (size_t)(t - 1) * H_, T_ * H_, enc_Whh, H_, 1, nullptr,
                 gates, 4 * H_, B_, 4 * H_, H_, 1, 0);
        lstm_enc_cell<<<(nBH + 255) / 256, 256, 0, stream>>>(
            gates, cstate, enc + (size_t)t * H_, T_ * H_, nBH);
    }

    // ---- whs = w-MLP(enc), loop-invariant ----
    gemm(stream, enc,  H_, w_Ws + 0 * H_ * H_, H_, 0, w_bs + 0 * H_, buft, H_, (int)BT, H_, H_, 0, 1);
    gemm(stream, buft, H_, w_Ws + 1 * H_ * H_, H_, 0, w_bs + 1 * H_, bufh, H_, (int)BT, H_, H_, 0, 1);
    gemm(stream, bufh, H_, w_Ws + 2 * H_ * H_, H_, 0, w_bs + 2 * H_, buft, H_, (int)BT, H_, H_, 0, 1);
    gemm(stream, buft, H_, w_Wl,               H_, 0, w_bl,          whs,  H_, (int)BT, H_, H_, 0, 0);

    // ---- autoregressive decode (OUT_LEN steps) ----
    const int zBlocks = (int)((BT * H_) / 256);
    for (int step = 0; step < OUTL_; ++step) {
        // us = u-MLP(s)
        gemm(stream, sstate, H_, u_Ws + 0 * H_ * H_, H_, 0, u_bs + 0 * H_, usa, H_, B_, H_, H_, 0, 1);
        gemm(stream, usa,    H_, u_Ws + 1 * H_ * H_, H_, 0, u_bs + 1 * H_, usb, H_, B_, H_, H_, 0, 1);
        gemm(stream, usb,    H_, u_Ws + 2 * H_ * H_, H_, 0, u_bs + 2 * H_, usa, H_, B_, H_, H_, 0, 1);
        gemm(stream, usa,    H_, u_Wl,               H_, 0, u_bl,          usb, H_, B_, H_, H_, 0, 0);

        // z = tanh(us[:,None,:] + whs)
        add_tanh<<<zBlocks, 256, 0, stream>>>(usb, whs, buft);

        // v-MLP: 7 hidden layers ping-ponging buft <-> bufh (ends in bufh)
        float* ping = buft; float* pong = bufh;
        for (int k = 0; k < 7; ++k) {
            gemm(stream, ping, H_, v_Ws + (size_t)k * H_ * H_, H_, 0, v_bs + k * H_,
                 pong, H_, (int)BT, H_, H_, 0, 1);
            float* tmp = ping; ping = pong; pong = tmp;
        }
        // final N=1 layer: e = z @ v_Wl + v_bl
        rowdot<<<(int)BT, 256, 0, stream>>>(ping, v_Wl, v_bl, evec);

        // attention softmax over time, context vector
        softmax_T<<<B_, 256, 0, stream>>>(evec, avec);
        ctx_sum<<<B_, 256, 0, stream>>>(avec, enc, ct);

        // dec gates = [c_t, s] @ dec_Wih^T + (bih+bhh), split along K
        gemm(stream, ct,     H_, dec_Wih,      2 * H_, 1, bsum_d, gates, 4 * H_, B_, 4 * H_, H_, 0, 0);
        gemm(stream, sstate, H_, dec_Wih + H_, 2 * H_, 1, nullptr, gates, 4 * H_, B_, 4 * H_, H_, 1, 0);
        lstm_dec_cell<<<(nBH + 255) / 256, 256, 0, stream>>>(gates, sstate, decseq, step, nBH);
    }

    // ---- output projection: out = dec @ out_W + out_b  [4096,256]x[256,64] ----
    gemm(stream, decseq, H_, out_W, DOUT_, 0, out_b, out, DOUT_, B_ * OUTL_, DOUT_, H_, 0, 0);
}
